// MoELayer_1571958030853
// MI455X (gfx1250) — matrix operand order repaired and checked by reading, compiled, not run on hardware
//
#include <hip/hip_runtime.h>
#include <math.h>

// Problem constants (from the reference): B=4, S=2048, D=1024, H=4096, DO=1024, E=8, K=2
#define T_TOK 8192
#define D_IN  1024
#define H_HID 4096
#define D_OUT 1024
#define NEXP  8
#define TILE_M 32               // tokens per workgroup tile (2x B-operand reuse)
#define CHUNK 512               // hidden-dim chunk kept in LDS
#define XP (D_IN + 8)           // padded LDS row stride (halfs) for x tile
#define HP (CHUNK + 8)          // padded LDS row stride (halfs) for h tile
#define XS_BYTES ((size_t)TILE_M * XP * 2)   // 66048
#define HS_BYTES ((size_t)TILE_M * HP * 2)   // 33280
#define SMEM_BYTES (XS_BYTES + HS_BYTES + TILE_M * 4 + TILE_M * 4)  // ~97.3 KB dynamic LDS

typedef __attribute__((ext_vector_type(16))) _Float16 v16h;
typedef __attribute__((ext_vector_type(8)))  float    v8f;

// ---------------- init: zero out+weights output regions and counters ----------------
__global__ void init_kernel(float* __restrict__ out, int* __restrict__ cnt) {
    size_t i = (size_t)blockIdx.x * 256 + threadIdx.x;
    const size_t nzero = (size_t)T_TOK * D_OUT + (size_t)T_TOK * NEXP; // out + weights
    if (i < nzero) out[i] = 0.0f;
    if (i < NEXP) cnt[i] = 0;
}

// ------------- convert w1 [e][d][h] f32 -> w1h [e][h][d] f16 (B-operand friendly) ----
__global__ void cvt_w1_kernel(const float* __restrict__ w1, _Float16* __restrict__ w1h) {
    size_t i = (size_t)blockIdx.x * 256 + threadIdx.x;
    if (i >= (size_t)NEXP * D_IN * H_HID) return;
    int h = (int)(i % H_HID);
    size_t r = i / H_HID;
    int d = (int)(r % D_IN);
    int e = (int)(r / D_IN);
    w1h[((size_t)e * H_HID + h) * D_IN + d] = (_Float16)w1[i];
}

// ------------- convert w2 [e][h][o] f32 -> w2h [e][o][h] f16 -------------------------
__global__ void cvt_w2_kernel(const float* __restrict__ w2, _Float16* __restrict__ w2h) {
    size_t i = (size_t)blockIdx.x * 256 + threadIdx.x;
    if (i >= (size_t)NEXP * H_HID * D_OUT) return;
    int o = (int)(i % D_OUT);
    size_t r = i / D_OUT;
    int h = (int)(r % H_HID);
    int e = (int)(r / H_HID);
    w2h[((size_t)e * D_OUT + o) * H_HID + h] = (_Float16)w2[i];
}

// ------------- gate: logits, top-2, softmax, routing lists ---------------------------
__global__ void gate_kernel(const float* __restrict__ x,
                            const float* __restrict__ gw,
                            const float* __restrict__ gb,
                            float* __restrict__ weights_out,
                            float* __restrict__ topi_out,
                            int*   __restrict__ cnt,
                            int*   __restrict__ ltok,
                            float* __restrict__ lw) {
    const int lane = threadIdx.x & 31;
    const int t = blockIdx.x * 8 + (threadIdx.x >> 5);   // one wave32 per token
    float acc[NEXP];
#pragma unroll
    for (int e = 0; e < NEXP; ++e) acc[e] = 0.0f;

    const float* xr = x + (size_t)t * D_IN;
    for (int d = lane; d < D_IN; d += 32) {
        float xv = xr[d];
        const float4* g4 = (const float4*)(gw + (size_t)d * NEXP);
        float4 a = g4[0], b = g4[1];
        acc[0] += xv * a.x; acc[1] += xv * a.y; acc[2] += xv * a.z; acc[3] += xv * a.w;
        acc[4] += xv * b.x; acc[5] += xv * b.y; acc[6] += xv * b.z; acc[7] += xv * b.w;
    }
#pragma unroll
    for (int e = 0; e < NEXP; ++e) {
#pragma unroll
        for (int off = 16; off > 0; off >>= 1) acc[e] += __shfl_xor(acc[e], off, 32);
    }
    if (lane == 0) {
        float l[NEXP];
#pragma unroll
        for (int e = 0; e < NEXP; ++e) l[e] = acc[e] + gb[e];
        int e0 = 0; float v0 = l[0];
#pragma unroll
        for (int e = 1; e < NEXP; ++e) { if (l[e] > v0) { v0 = l[e]; e0 = e; } }
        int e1 = -1; float v1 = -3.4e38f;
#pragma unroll
        for (int e = 0; e < NEXP; ++e) { if (e != e0 && l[e] > v1) { v1 = l[e]; e1 = e; } }
        // softmax over [v0, v1] (v0 >= v1)
        float g0 = 1.0f / (1.0f + __expf(v1 - v0));
        float g1 = 1.0f - g0;
        weights_out[t * NEXP + e0] = g0;
        weights_out[t * NEXP + e1] = g1;
        topi_out[t * 2 + 0] = (float)e0;
        topi_out[t * 2 + 1] = (float)e1;
        int s0 = atomicAdd(&cnt[e0], 1);
        ltok[e0 * T_TOK + s0] = t; lw[e0 * T_TOK + s0] = g0;
        int s1 = atomicAdd(&cnt[e1], 1);
        ltok[e1 * T_TOK + s1] = t; lw[e1 * T_TOK + s1] = g1;
    }
}

// ------------- fused expert MLP: 32-token tile, two chained WMMA GEMMs ---------------
// Each B-operand fragment (32x16 f16 weight tile) feeds TWO WMMAs (row tiles 0-15 and
// 16-31), doubling L2 arithmetic intensity vs a 16-token tile.
__launch_bounds__(256)
__global__ void moe_expert_kernel(const float* __restrict__ x,
                                  const _Float16* __restrict__ w1h,
                                  const float* __restrict__ b1,
                                  const _Float16* __restrict__ w2h,
                                  const float* __restrict__ b2,
                                  const int* __restrict__ cnt,
                                  const int* __restrict__ ltok,
                                  const float* __restrict__ lw,
                                  float* __restrict__ out) {
    extern __shared__ char smem[];
    _Float16* xs = (_Float16*)smem;                       // [TILE_M][XP]
    _Float16* hs = (_Float16*)(smem + XS_BYTES);          // [TILE_M][HP]
    float*   gsh = (float*)(smem + XS_BYTES + HS_BYTES);  // [TILE_M]
    int*     tsh = (int*)(gsh + TILE_M);                  // [TILE_M]

    const int e = blockIdx.y;
    const int tile = blockIdx.x;
    const int n_e = cnt[e];
    if (tile * TILE_M >= n_e) return;

    const int tid  = threadIdx.x;
    const int lane = tid & 31;
    const int wv   = tid >> 5;       // 8 waves
    const int half = lane >> 4;      // 0: lanes 0-15, 1: lanes 16-31
    const int l15  = lane & 15;

    if (tid < TILE_M) {
        int slot = tile * TILE_M + tid;
        if (slot < n_e) { tsh[tid] = ltok[e * T_TOK + slot]; gsh[tid] = lw[e * T_TOK + slot]; }
        else            { tsh[tid] = -1;                     gsh[tid] = 0.0f; }
    }
    __syncthreads();

    // stage x tile -> LDS as fp16
    for (int i = tid; i < TILE_M * D_IN; i += 256) {
        int m = i >> 10;            // / D_IN
        int k = i & (D_IN - 1);
        int t = tsh[m];
        xs[m * XP + k] = (t >= 0) ? (_Float16)x[(size_t)t * D_IN + k] : (_Float16)0.0f;
    }
    __syncthreads();

    // phase-2 accumulators persist across hidden chunks:
    // wave covers out cols [wv*128, wv*128+128), both 16-row tiles => 128 f32 regs/lane
    v8f acc2[8][2];
#pragma unroll
    for (int nt = 0; nt < 8; ++nt) {
        float bv = b2[e * D_OUT + wv * 128 + nt * 16 + l15];
#pragma unroll
        for (int r = 0; r < 8; ++r) { acc2[nt][0][r] = bv; acc2[nt][1][r] = bv; }
    }

    for (int hc = 0; hc < H_HID / CHUNK; ++hc) {
        // ---- phase 1: h[32 x CHUNK] = relu(x @ w1 + b1), wave covers 64 cols ----
        for (int nt = 0; nt < 4; ++nt) {
            const int ncl = wv * 64 + nt * 16 + l15;       // col within chunk
            const int ng  = hc * CHUNK + ncl;              // global hidden col
            float bv = b1[e * H_HID + ng];
            v8f acc0, acc1;
#pragma unroll
            for (int r = 0; r < 8; ++r) { acc0[r] = bv; acc1[r] = bv; }
            const _Float16* bptr = w1h + ((size_t)e * H_HID + ng) * D_IN + half * 16;
#pragma unroll 2
            for (int kt = 0; kt < D_IN / 32; ++kt) {
                union { v16h v; float4 f[2]; } a0, a1;
                const int k0 = kt * 32 + half * 16;        // contiguous k = 16h + s (matches B)
                a0.f[0] = *(const float4*)&xs[l15 * XP + k0];
                a0.f[1] = *(const float4*)&xs[l15 * XP + k0 + 8];
                a1.f[0] = *(const float4*)&xs[(l15 + 16) * XP + k0];
                a1.f[1] = *(const float4*)&xs[(l15 + 16) * XP + k0 + 8];
                v16h bvv = *(const v16h*)(bptr + kt * 32);
                __builtin_prefetch(bptr + kt * 32 + 256, 0, 3);
                acc0 = __builtin_amdgcn_wmma_f32_16x16x32_f16(
                        false, a0.v, false, bvv, (short)0, acc0, false, false);
                acc1 = __builtin_amdgcn_wmma_f32_16x16x32_f16(
                        false, a1.v, false, bvv, (short)0, acc1, false, false);
            }
#pragma unroll
            for (int r = 0; r < 8; ++r) {                  // relu + fp16 store
                int m = r + half * 8;
                float v0 = acc0[r] > 0.0f ? acc0[r] : 0.0f;
                float v1 = acc1[r] > 0.0f ? acc1[r] : 0.0f;
                hs[m * HP + ncl] = (_Float16)v0;
                hs[(m + 16) * HP + ncl] = (_Float16)v1;
            }
        }
        __syncthreads();

        // ---- phase 2 partial: acc2 += h_chunk @ w2_chunk, wave covers 128 out cols ----
#pragma unroll
        for (int nt = 0; nt < 8; ++nt) {
            const int no = wv * 128 + nt * 16 + l15;
            const _Float16* bptr = w2h + ((size_t)e * D_OUT + no) * H_HID + hc * CHUNK + half * 16;
            v8f acc0 = acc2[nt][0];
            v8f acc1 = acc2[nt][1];
#pragma unroll 2
            for (int kt = 0; kt < CHUNK / 32; ++kt) {
                union { v16h v; float4 f[2]; } a0, a1;
                const int k0 = kt * 32 + half * 16;
                a0.f[0] = *(const float4*)&hs[l15 * HP + k0];
                a0.f[1] = *(const float4*)&hs[l15 * HP + k0 + 8];
                a1.f[0] = *(const float4*)&hs[(l15 + 16) * HP + k0];
                a1.f[1] = *(const float4*)&hs[(l15 + 16) * HP + k0 + 8];
                v16h bvv = *(const v16h*)(bptr + kt * 32);
                acc0 = __builtin_amdgcn_wmma_f32_16x16x32_f16(
                        false, a0.v, false, bvv, (short)0, acc0, false, false);
                acc1 = __builtin_amdgcn_wmma_f32_16x16x32_f16(
                        false, a1.v, false, bvv, (short)0, acc1, false, false);
            }
            acc2[nt][0] = acc0;
            acc2[nt][1] = acc1;
        }
        __syncthreads();
    }

    // gate-weighted scatter-accumulate into out (each token has 2 expert contributions)
#pragma unroll
    for (int nt = 0; nt < 8; ++nt) {
        const int no = wv * 128 + nt * 16 + l15;
#pragma unroll
        for (int r = 0; r < 8; ++r) {
            int m = r + half * 8;
            int t0 = tsh[m];
            int t1 = tsh[m + 16];
            if (t0 >= 0) atomicAdd(&out[(size_t)t0 * D_OUT + no], acc2[nt][0][r] * gsh[m]);
            if (t1 >= 0) atomicAdd(&out[(size_t)t1 * D_OUT + no], acc2[nt][1][r] * gsh[m + 16]);
        }
    }
}

// ------------- expert usage counts -> output ----------------------------------------
__global__ void usage_kernel(const int* __restrict__ cnt, float* __restrict__ usage_out) {
    if (threadIdx.x < NEXP) usage_out[threadIdx.x] = (float)cnt[threadIdx.x];
}

extern "C" void kernel_launch(void* const* d_in, const int* in_sizes, int n_in,
                              void* d_out, int out_size, void* d_ws, size_t ws_size,
                              hipStream_t stream) {
    const float* x      = (const float*)d_in[0];
    const float* gate_w = (const float*)d_in[1];
    const float* gate_b = (const float*)d_in[2];
    const float* w1     = (const float*)d_in[3];
    const float* b1     = (const float*)d_in[4];
    const float* w2     = (const float*)d_in[5];
    const float* b2     = (const float*)d_in[6];

    float* out         = (float*)d_out;                           // [T, DO]
    float* weights_out = out + (size_t)T_TOK * D_OUT;             // [T, E]
    float* usage_out   = weights_out + (size_t)T_TOK * NEXP;      // [E]
    float* topi_out    = usage_out + NEXP;                        // [T, K]

    char* ws = (char*)d_ws;
    _Float16* w1h = (_Float16*)ws;                                         // 64 MB
    _Float16* w2h = (_Float16*)(ws + (size_t)NEXP * D_IN * H_HID * 2);     // 64 MB
    char* p = ws + (size_t)NEXP * D_IN * H_HID * 2 + (size_t)NEXP * H_HID * D_OUT * 2;
    int*   cnt  = (int*)p;                                                 // [E]
    int*   ltok = (int*)(p + 64);                                          // [E, T]
    float* lw   = (float*)(p + 64 + sizeof(int) * (size_t)NEXP * T_TOK);   // [E, T]

    const size_t nzero = (size_t)T_TOK * D_OUT + (size_t)T_TOK * NEXP;
    init_kernel<<<(unsigned)((nzero + 255) / 256), 256, 0, stream>>>(out, cnt);

    const size_t nw = (size_t)NEXP * D_IN * H_HID;
    cvt_w1_kernel<<<(unsigned)((nw + 255) / 256), 256, 0, stream>>>(w1, w1h);
    cvt_w2_kernel<<<(unsigned)((nw + 255) / 256), 256, 0, stream>>>(w2, w2h);

    gate_kernel<<<T_TOK / 8, 256, 0, stream>>>(x, gate_w, gate_b,
                                               weights_out, topi_out, cnt, ltok, lw);

    moe_expert_kernel<<<dim3(T_TOK / TILE_M, NEXP), 256, (unsigned)SMEM_BYTES, stream>>>(
        x, w1h, b1, w2h, b2, cnt, ltok, lw, out);

    usage_kernel<<<1, 32, 0, stream>>>(cnt, usage_out);
}
